// NeuralEnergyLandscape_35003983462986
// MI455X (gfx1250) — matrix operand order repaired and checked by reading, not compiled, not run on hardware
//
#include <hip/hip_runtime.h>
#include <hip/hip_bf16.h>

typedef __attribute__((ext_vector_type(16))) _Float16 v16h;
typedef __attribute__((ext_vector_type(8)))  float    v8f;

// ---------------- LDS layout (bytes) ----------------
#define OFF_W1    0        // 128 cols x 64 k halfs   (16384)
#define OFF_W2    16384    // 64 cols x 128 k halfs   (16384)
#define OFF_W3    32768    // 16 cols x 64 k halfs    (2048)
#define OFF_B1    34816    // 128 f32
#define OFF_B2    35328    // 64 f32
#define OFF_RCPM  35584    // 14 f32 (padded)
#define OFF_PRIOR 35648    // 64 f32
#define OFF_SUSP  35904    // 64 f32
#define OFF_SCR1  36160    // act1 (64x128 halfs = 16KB) ; also raw q/p/setup before barrier
#define OFF_QB    36160    // 64*14 f32 = 3584
#define OFF_PB    39744    // 64*14 f32 = 3584
#define OFF_SB    43328    // 64*28 f32 = 7168
#define OFF_SCR2  52544    // feat (64x64 halfs = 8KB) ; later act2 (64x64 halfs)
#define LDS_BYTES 60736

#define PERSISTENT_BLOCKS 2048

union Frag { v16h h; unsigned int u[8]; };

// A fragment, 16x32 f16 (M = lane&15). lanes 0-15: K 0-15 ; lanes 16-31: K 16-31
__device__ __forceinline__ v16h load_a_frag(const _Float16* row, int kbase, int lane) {
  Frag f;
  const unsigned char* bp =
      (const unsigned char*)row + 2u * (unsigned)(kbase + ((lane & 16) ? 16 : 0));
#pragma unroll
  for (int v = 0; v < 8; ++v) {
    const int koff = 2 * v;
    f.u[v] = *(const unsigned int*)(bp + 2 * koff);
  }
  return f.h;
}

// B fragment, 32x16 f16 (N = lane&15). lanes 0-15: K 0-15 ; lanes 16-31: K 16-31. col = K-major in LDS
__device__ __forceinline__ v16h load_b_frag(const _Float16* col, int kbase, int lane) {
  Frag f;
  const unsigned char* bp =
      (const unsigned char*)col + 2u * (unsigned)(kbase + ((lane & 16) ? 16 : 0));
#pragma unroll
  for (int v = 0; v < 8; ++v) f.u[v] = *(const unsigned int*)(bp + 4 * v);
  return f.h;
}

#define WMMA_F16(A, Bf, C) \
  __builtin_amdgcn_wmma_f32_16x16x32_f16(false, (A), false, (Bf), (short)0, (C), false, false)

// swish via raw v_rcp_f32 (no IEEE div chain) -- plenty of precision next to f16 WMMA
__device__ __forceinline__ float swish_f(float x) {
  return x * __builtin_amdgcn_rcpf(1.0f + __expf(-x));
}

#define RS(s)  (1.0f / ((s) + 1e-6f))
#define RS2(s) (1.0f / ((s) * (s) + 1e-6f))
#define SF(i, m, s) ((ss[(i)] - (m)) * RS(s))

__global__ __launch_bounds__(128) void nel_kernel(
    const float* __restrict__ q, const float* __restrict__ p, const float* __restrict__ sp,
    const float* __restrict__ Md, const float* __restrict__ W1, const float* __restrict__ b1,
    const float* __restrict__ W2, const float* __restrict__ b2, const float* __restrict__ W3,
    const float* __restrict__ b3, float* __restrict__ out, int B, int nTiles) {
  __shared__ __align__(16) unsigned char smem[LDS_BYTES];
  _Float16* w1l   = (_Float16*)(smem + OFF_W1);
  _Float16* w2l   = (_Float16*)(smem + OFF_W2);
  _Float16* w3l   = (_Float16*)(smem + OFF_W3);
  float*    b1l   = (float*)(smem + OFF_B1);
  float*    b2l   = (float*)(smem + OFF_B2);
  float*    rMl   = (float*)(smem + OFF_RCPM);
  float*    priorl = (float*)(smem + OFF_PRIOR);
  float*    suspl  = (float*)(smem + OFF_SUSP);
  float*    qb = (float*)(smem + OFF_QB);
  float*    pb = (float*)(smem + OFF_PB);
  float*    sb = (float*)(smem + OFF_SB);
  _Float16* act1 = (_Float16*)(smem + OFF_SCR1);
  _Float16* feat = (_Float16*)(smem + OFF_SCR2);
  _Float16* act2 = (_Float16*)(smem + OFF_SCR2);

  const int tid  = threadIdx.x;
  const int lane = tid & 31;
  const int wave = tid >> 5;
  const long long qlim = (long long)B * 14 - 1;
  const long long slim = (long long)B * 28 - 1;
  const int step = (int)gridDim.x;

  // ---- stage weights to LDS as f16, column(K)-major: ONCE per persistent block ----
  for (int i = tid; i < 8192; i += 128) {            // W1: [n][k], k padded 38->64
    const int n = i >> 6, k = i & 63;
    w1l[i] = (_Float16)((k < 38) ? W1[k * 128 + n] : 0.0f);
  }
  for (int i = tid; i < 8192; i += 128) {            // W2: [n][k], k=0..127
    const int n = i >> 7, k = i & 127;
    w2l[i] = (_Float16)W2[k * 64 + n];
  }
  for (int i = tid; i < 1024; i += 128) {            // W3: [n][k], only n==0 non-zero
    const int n = i >> 6, k = i & 63;
    w3l[i] = (_Float16)((n == 0) ? W3[k] : 0.0f);
  }
  b1l[tid] = b1[tid];
  if (tid < 64) b2l[tid] = b2[tid];
  if (tid < 14) rMl[tid] = 1.0f / Md[tid];
  const float bb3 = b3[0];

  // ---- persistent grid-stride loop over 64-sample tiles ----
  for (int tile = (int)blockIdx.x; tile < nTiles; tile += step) {
    const long long base = (long long)tile * 64;

    // barrier: previous iteration's act1/act2 LDS use must finish before re-staging raw inputs
    __syncthreads();

    // prefetch this block's NEXT tile (far enough ahead to be useful in L2)
    if (tile + step < nTiles) {
      const long long nbase = (long long)(tile + step) * 64;
      __builtin_prefetch(q  + nbase * 14 + tid * 8, 0, 0);
      __builtin_prefetch(p  + nbase * 14 + tid * 8, 0, 0);
      __builtin_prefetch(sp + nbase * 28 + tid * 16, 0, 0);
    }

    // ---- stage raw inputs coalesced ----
    for (int i = tid; i < 64 * 14; i += 128) {
      long long idx = base * 14 + i; if (idx > qlim) idx = qlim;
      qb[i] = q[idx];
    }
    for (int i = tid; i < 64 * 14; i += 128) {
      long long idx = base * 14 + i; if (idx > qlim) idx = qlim;
      pb[i] = p[idx];
    }
    for (int i = tid; i < 64 * 28; i += 128) {
      long long idx = base * 28 + i; if (idx > slim) idx = slim;
      sb[i] = sp[idx];
    }
    __syncthreads();

    // ---- features (one thread per sample; waves 0-1 only, wave-uniform branch) ----
    if (tid < 64) {
      const int t = tid;
      const float* qq = qb + t * 14;
      const float* pp = pb + t * 14;
      const float* ss = sb + t * 28;
      float v[14];
      float Tp = 0.0f;
#pragma unroll
      for (int i = 0; i < 14; ++i) {
        v[i] = pp[i] * rMl[i];
        Tp += pp[i] * v[i];
      }
      Tp *= 0.5f;
      const float susp = qq[6]*qq[6] + qq[7]*qq[7] + qq[8]*qq[8] + qq[9]*qq[9];
      priorl[t] = Tp + 0.5f * susp * 30000.0f;
      suspl[t]  = susp;

      float f[38];
      // sym_q_norm
      f[0] = qq[2] * RS(0.1f);
      f[1] = qq[4] * RS(0.1f);
      f[2] = (qq[6] + qq[7]) * 0.5f * RS(0.05f);
      f[3] = (qq[8] + qq[9]) * 0.5f * RS(0.05f);
      // sym_v_norm
      f[4]  = v[0] * RS(25.0f);
      f[5]  = v[2] * RS(1.0f);
      f[6]  = v[4] * RS(1.5f);
      f[7]  = (v[6]  + v[7])  * 0.5f * RS(1.0f);
      f[8]  = (v[8]  + v[9])  * 0.5f * RS(1.0f);
      f[9]  = (v[10] + v[11]) * 0.5f * RS(75.0f);
      f[10] = (v[12] + v[13]) * 0.5f * RS(75.0f);
      // anti_q_norm (squared)
      const float d1 = qq[6] - qq[7];
      const float d2 = qq[8] - qq[9];
      const float d3 = qq[6] + qq[9] - qq[7] - qq[8];
      f[11] = qq[3]*qq[3] * RS2(0.1f);
      f[12] = d1*d1 * RS2(0.05f);
      f[13] = d2*d2 * RS2(0.05f);
      f[14] = d3*d3 * RS2(0.05f);
      // anti_v_norm (squared)
      const float e1 = v[6]  - v[7];
      const float e2 = v[8]  - v[9];
      const float e3 = v[10] - v[11];
      const float e4 = v[12] - v[13];
      f[15] = v[1]*v[1] * RS2(5.0f);
      f[16] = v[3]*v[3] * RS2(1.0f);
      f[17] = v[5]*v[5] * RS2(1.0f);
      f[18] = e1*e1 * RS2(75.0f);
      f[19] = e2*e2 * RS2(75.0f);
      f[20] = e3*e3 * RS2(75.0f);
      f[21] = e4*e4 * RS2(75.0f);
      // setup_feat, SETUP_FEAT_IDX = [0,1,2,3,4,5,14,15,12,13,25,19,20,21,22,24]
      f[22] = SF(0,  40000.0f, 30000.0f);
      f[23] = SF(1,  40000.0f, 30000.0f);
      f[24] = SF(2,  500.0f,   500.0f);
      f[25] = SF(3,  500.0f,   500.0f);
      f[26] = SF(4,  3000.0f,  2000.0f);
      f[27] = SF(5,  3000.0f,  2000.0f);
      f[28] = SF(14, -2.0f,    2.0f);
      f[29] = SF(15, -1.5f,    2.0f);
      f[30] = SF(12, 0.03f,    0.02f);
      f[31] = SF(13, 0.03f,    0.02f);
      f[32] = SF(25, 0.285f,   0.05f);
      f[33] = SF(19, 0.35f,    0.25f);
      f[34] = SF(20, 0.35f,    0.25f);
      f[35] = SF(21, 0.15f,    0.2f);
      f[36] = SF(22, 0.25f,    0.2f);
      f[37] = SF(24, 0.6f,     0.15f);

      _Float16* fr = feat + t * 64;
#pragma unroll
      for (int i = 0; i < 38; ++i) fr[i] = (_Float16)f[i];
#pragma unroll
      for (int i = 38; i < 64; ++i) fr[i] = (_Float16)0.0f;
    }
    __syncthreads();

    // ---- per-wave WMMA MLP: 16 samples per wave ----
    const int s0    = wave * 16;
    const int nlo   = lane & 15;
    const int mbase = (lane & 16) ? 8 : 0;

    // Layer 1: [16x64] x [64x128] -> 16x128, swish
    const _Float16* arow1 = feat + (s0 + nlo) * 64;
    const v16h a0 = load_a_frag(arow1, 0, lane);
    const v16h a1 = load_a_frag(arow1, 32, lane);
    _Float16* act1w = act1 + s0 * 128;
#pragma unroll
    for (int nt = 0; nt < 8; ++nt) {
      const _Float16* col = w1l + (nt * 16 + nlo) * 64;
      v8f c = {};
      c = WMMA_F16(a0, load_b_frag(col, 0, lane), c);
      c = WMMA_F16(a1, load_b_frag(col, 32, lane), c);
      const float bias = b1l[nt * 16 + nlo];
#pragma unroll
      for (int r = 0; r < 8; ++r) {
        const float x = c[r] + bias;
        act1w[(mbase + r) * 128 + nt * 16 + nlo] = (_Float16)swish_f(x);
      }
    }

    // Layer 2: [16x128] x [128x64] -> 16x64, swish
    const _Float16* arow2 = act1 + (s0 + nlo) * 128;
    v16h a2[4];
#pragma unroll
    for (int ks = 0; ks < 4; ++ks) a2[ks] = load_a_frag(arow2, ks * 32, lane);
    _Float16* act2w = act2 + s0 * 64;
#pragma unroll
    for (int nt = 0; nt < 4; ++nt) {
      const _Float16* col = w2l + (nt * 16 + nlo) * 128;
      v8f c = {};
#pragma unroll
      for (int ks = 0; ks < 4; ++ks) c = WMMA_F16(a2[ks], load_b_frag(col, ks * 32, lane), c);
      const float bias = b2l[nt * 16 + nlo];
#pragma unroll
      for (int r = 0; r < 8; ++r) {
        const float x = c[r] + bias;
        act2w[(mbase + r) * 64 + nt * 16 + nlo] = (_Float16)swish_f(x);
      }
    }

    // Layer 3: [16x64] x [64x16] -> 16x16 (only N=0 meaningful)
    const _Float16* arow3 = act2 + (s0 + nlo) * 64;
    const v16h a30 = load_a_frag(arow3, 0, lane);
    const v16h a31 = load_a_frag(arow3, 32, lane);
    const _Float16* col3 = w3l + nlo * 64;
    v8f c3 = {};
    c3 = WMMA_F16(a30, load_b_frag(col3, 0, lane), c3);
    c3 = WMMA_F16(a31, load_b_frag(col3, 32, lane), c3);

    // ---- epilogue: softplus cap + physical terms; only lanes with N==0 hold column 0 ----
    if (nlo == 0) {
#pragma unroll
      for (int r = 0; r < 8; ++r) {
        const int sl = s0 + mbase + r;
        const long long gS = base + sl;
        const float y  = c3[r] + bb3;
        float spv = (y > 20.0f) ? y : __logf(1.0f + __expf(y));
        spv = fminf(spv, 5000.0f);
        if (gS < (long long)B) out[gS] = priorl[sl] + spv * suspl[sl];
      }
    }
  }
}

extern "C" void kernel_launch(void* const* d_in, const int* in_sizes, int n_in,
                              void* d_out, int out_size, void* d_ws, size_t ws_size,
                              hipStream_t stream) {
  const float* q  = (const float*)d_in[0];
  const float* p  = (const float*)d_in[1];
  const float* sp = (const float*)d_in[2];
  const float* Md = (const float*)d_in[3];
  const float* W1 = (const float*)d_in[4];
  const float* b1 = (const float*)d_in[5];
  const float* W2 = (const float*)d_in[6];
  const float* b2 = (const float*)d_in[7];
  const float* W3 = (const float*)d_in[8];
  const float* b3 = (const float*)d_in[9];
  float* out = (float*)d_out;
  const int B = in_sizes[0] / 14;
  const int nTiles = (B + 63) / 64;
  const int grid = (nTiles < PERSISTENT_BLOCKS) ? nTiles : PERSISTENT_BLOCKS;
  hipLaunchKernelGGL(nel_kernel, dim3(grid), dim3(128), 0, stream,
                     q, p, sp, Md, W1, b1, W2, b2, W3, b3, out, B, nTiles);
}
